// LSTM_55370718380408
// MI455X (gfx1250) — matrix operand order repaired and checked by reading, not compiled, not run on hardware
//
#include <hip/hip_runtime.h>
#include <stdint.h>

// ---------------------------------------------------------------------------
// LSTM for MI455X (gfx1250): bf16 WMMA (16x16x32, f32 accum), wave32.
//   Kernel 1: transpose+convert weights to bf16 [N][K] layout (workspace)
//   Kernel 2: xg = x @ W_all + b_all (parallel WMMA GEMM, bf16 output,
//             biases folded in so the scan loop never touches them)
//   Kernel 3: persistent per-batch-group scan: h @ U_all via register-resident
//             bf16 U fragments + LDS gate buffer; xg slab double-buffered into
//             LDS with global_load_async_to_lds (ASYNCcnt); fused h_T @ Why.
// ---------------------------------------------------------------------------

#define LSTM_B 64
#define LSTM_S 4096
#define LSTM_H 256
#define LSTM_G 1024 // 4H, gate order [f, i, o, c]

typedef __attribute__((ext_vector_type(16))) __bf16 v16bf;
typedef __attribute__((ext_vector_type(4)))  __bf16 v4bf;
typedef __attribute__((ext_vector_type(8)))  float  v8f;
typedef __attribute__((ext_vector_type(4)))  float  f32x4;
typedef __attribute__((ext_vector_type(4)))  unsigned int u32x4;
typedef __attribute__((ext_vector_type(2)))  int     i32x2;

union FragU { u32x4 u[2]; v16bf v; };

#if defined(__gfx1250__) && __has_builtin(__builtin_amdgcn_global_load_async_to_lds_b64)
#define LSTM_ASYNC 1
#else
#define LSTM_ASYNC 0
#endif

__device__ __forceinline__ float fast_sigmoid(float x) {
  float e = __builtin_amdgcn_exp2f(-1.4426950408889634f * x);
  return __builtin_amdgcn_rcpf(1.0f + e);
}
__device__ __forceinline__ float fast_tanh(float x) {
#if __has_builtin(__builtin_amdgcn_tanhf)
  return __builtin_amdgcn_tanhf(x);
#else
  float e = __builtin_amdgcn_exp2f(2.8853900817779268f * x);
  return 1.0f - 2.0f * __builtin_amdgcn_rcpf(e + 1.0f);
#endif
}

#if LSTM_ASYNC
typedef __attribute__((address_space(1))) i32x2 gas_i32x2;
typedef __attribute__((address_space(3))) i32x2 las_i32x2;

__device__ __forceinline__ void async_ld_b64(const __bf16* g, __bf16* lds) {
  __builtin_amdgcn_global_load_async_to_lds_b64(
      (gas_i32x2*)g, (las_i32x2*)lds, 0, 0);
}
__device__ __forceinline__ void wait_async4() {
#if __has_builtin(__builtin_amdgcn_s_wait_asynccnt)
  __builtin_amdgcn_s_wait_asynccnt(4);
#else
  asm volatile("s_wait_asynccnt 4" ::: "memory");
#endif
}
__device__ __forceinline__ void wait_async0() {
#if __has_builtin(__builtin_amdgcn_s_wait_asynccnt)
  __builtin_amdgcn_s_wait_asynccnt(0);
#else
  asm volatile("s_wait_asynccnt 0" ::: "memory");
#endif
}
#endif

// ---------------------------------------------------------------------------
// Kernel 1: build bf16 transposed weights in workspace.
//   Wt  [1024][256]  : Wt[g][k]  = W_all[k][g],  W_all cols = [Wf|Wi|Wo|Wc]
//   Utb [1024][256]  : Utb[g][k] = U_all[k][g]
//   Whyt [256][256]  : Whyt[n][k] = Why[k][n]
// ---------------------------------------------------------------------------
__global__ __launch_bounds__(256) void lstm_cvt_weights(
    const float* __restrict__ Wf, const float* __restrict__ Wi,
    const float* __restrict__ Wo, const float* __restrict__ Wc,
    const float* __restrict__ Uf, const float* __restrict__ Ui,
    const float* __restrict__ Uo, const float* __restrict__ Uc,
    const float* __restrict__ Why,
    __bf16* __restrict__ Wt, __bf16* __restrict__ Utb,
    __bf16* __restrict__ Whyt) {
  int row = blockIdx.x;
  int k = threadIdx.x; // 0..255
  if (row < 1024) {
    int gate = row >> 8, col = row & 255;
    const float* src = (gate == 0) ? Wf : (gate == 1) ? Wi : (gate == 2) ? Wo : Wc;
    Wt[(size_t)row * 256 + k] = (__bf16)src[k * 256 + col];
  } else if (row < 2048) {
    int r = row - 1024;
    int gate = r >> 8, col = r & 255;
    const float* src = (gate == 0) ? Uf : (gate == 1) ? Ui : (gate == 2) ? Uo : Uc;
    Utb[(size_t)r * 256 + k] = (__bf16)src[k * 256 + col];
  } else {
    int r = row - 2048;
    Whyt[(size_t)r * 256 + k] = (__bf16)Why[k * 256 + r];
  }
}

// ---------------------------------------------------------------------------
// Kernel 2: xg[m][g] = sum_k x_row(m)[k]*W_all[k][g] + b_all[g],  m = s*64+b.
// 256 threads = 8 waves/block; each wave: one M-tile x 4 N-tiles, K loop of 8.
// grid = 32768  (16384 M-tiles * 16 N-groups / 8 waves)
// ---------------------------------------------------------------------------
__global__ __launch_bounds__(256) void lstm_xproj(
    const float* __restrict__ x, const __bf16* __restrict__ Wt,
    const float* __restrict__ b_f, const float* __restrict__ b_i,
    const float* __restrict__ b_o, const float* __restrict__ b_c,
    __bf16* __restrict__ xg) {
  const int tid  = threadIdx.x;
  const int wid  = tid >> 5;
  const int lane = tid & 31;
  const int lrow = lane & 15;
  const int lhi  = lane >> 4;

  const int jid = blockIdx.x * 8 + wid;
  const int mt  = jid >> 4;   // 0..16383
  const int ng  = jid & 15;   // 0..15 -> 4 N-tiles each

  // A row m -> (s, b): m = s*64 + b ; x is [B][S][I]
  const int m0 = mt * 16 + lrow;
  const int s  = m0 >> 6;
  const int b  = m0 & 63;
  const float* xrow = x + ((size_t)b * LSTM_S + s) * LSTM_H;

  const v8f zero8 = {0.f, 0.f, 0.f, 0.f, 0.f, 0.f, 0.f, 0.f};
  v8f acc[4] = {zero8, zero8, zero8, zero8};

  #pragma unroll
  for (int kt = 0; kt < 8; ++kt) {
    // A fragment: lane holds row M=lrow, K = kt*32 + lhi*8 + {0..7, 16..23}
    const int k0 = kt * 32 + lhi * 8;
    f32x4 a0 = *(const f32x4*)(xrow + k0);
    f32x4 a1 = *(const f32x4*)(xrow + k0 + 4);
    f32x4 a2 = *(const f32x4*)(xrow + k0 + 16);
    f32x4 a3 = *(const f32x4*)(xrow + k0 + 20);
    v16bf av;
    av[0]=(__bf16)a0[0]; av[1]=(__bf16)a0[1]; av[2]=(__bf16)a0[2]; av[3]=(__bf16)a0[3];
    av[4]=(__bf16)a1[0]; av[5]=(__bf16)a1[1]; av[6]=(__bf16)a1[2]; av[7]=(__bf16)a1[3];
    av[8]=(__bf16)a2[0]; av[9]=(__bf16)a2[1]; av[10]=(__bf16)a2[2]; av[11]=(__bf16)a2[3];
    av[12]=(__bf16)a3[0]; av[13]=(__bf16)a3[1]; av[14]=(__bf16)a3[2]; av[15]=(__bf16)a3[3];

    #pragma unroll
    for (int j = 0; j < 4; ++j) {
      const int nt = ng * 4 + j;
      // B fragment: lane = col n, K = kt*32 + lhi*8 + {0..7, 16..23} in Wt row
      const __bf16* p = Wt + (size_t)(nt * 16 + lrow) * 256 + kt * 32 + lhi * 8;
      FragU bu;
      bu.u[0] = *(const u32x4*)(p);
      bu.u[1] = *(const u32x4*)(p + 16);
      acc[j] = __builtin_amdgcn_wmma_f32_16x16x32_bf16(
          false, av, false, bu.v, (short)0, acc[j], false, false);
    }
  }

  // D tile: VGPR r, lanes 0-15 -> M=r, lanes 16-31 -> M=r+8; col = lrow.
  // Fold gate bias (per column) into xg here.
  #pragma unroll
  for (int j = 0; j < 4; ++j) {
    const int gcol = (ng * 4 + j) * 16 + lrow;
    const int gate = gcol >> 8, col = gcol & 255;
    const float* bp = (gate == 0) ? b_f : (gate == 1) ? b_i : (gate == 2) ? b_o : b_c;
    const float bv = bp[col];
    #pragma unroll
    for (int r = 0; r < 8; ++r) {
      const int m = mt * 16 + r + lhi * 8;
      xg[(size_t)m * LSTM_G + gcol] = (__bf16)(acc[j][r] + bv);
    }
  }
}

// ---------------------------------------------------------------------------
// Kernel 3: recurrent scan. grid = 4 blocks (16 batch rows each), 1024 threads.
// ---------------------------------------------------------------------------
__global__ __launch_bounds__(1024) void lstm_scan(
    const __bf16* __restrict__ xg, const __bf16* __restrict__ Utb,
    const __bf16* __restrict__ Whyt,
    const float* __restrict__ b_y, float* __restrict__ out) {
  __shared__ __bf16 hbuf[16][264];   // 264-half pitch: conflict-free b128 reads
  __shared__ float  gbuf[16][1032];  // gate pre-activations g = h @ U_all
#if LSTM_ASYNC
  __shared__ __bf16 sbuf[2][16][1024];  // double-buffered xg slab (async-to-LDS)
#endif

  const int tid  = threadIdx.x;
  const int wid  = tid >> 5;   // 0..31
  const int lane = tid & 31;
  const int lrow = lane & 15;
  const int lhi  = lane >> 4;
  const int bbase = blockIdx.x * 16;

  // ---- preload register-resident U fragments (2 N-tiles x 8 K-tiles) ----
  v16bf Bf[2][8];
  #pragma unroll
  for (int t = 0; t < 2; ++t) {
    const int nt = wid * 2 + t;
    #pragma unroll
    for (int kt = 0; kt < 8; ++kt) {
      const __bf16* p = Utb + (size_t)(nt * 16 + lrow) * 256 + kt * 32 + lhi * 8;
      FragU fu;
      fu.u[0] = *(const u32x4*)(p);
      fu.u[1] = *(const u32x4*)(p + 16);
      Bf[t][kt] = fu.v;
    }
  }

  // ---- elementwise ownership: thread -> (batch row eb, 4 hidden cols ej..) --
  const int eb = tid >> 6;         // 0..15
  const int ej = (tid & 63) * 4;   // 0..252
  float Creg[4] = {0.f, 0.f, 0.f, 0.f};

  // zero-init h
  for (int i = tid; i < 16 * 264; i += 1024) ((__bf16*)hbuf)[i] = (__bf16)0.0f;

  const __bf16* xgrow = xg + (size_t)(bbase + eb) * LSTM_G + ej;
  const v8f zero8 = {0.f, 0.f, 0.f, 0.f, 0.f, 0.f, 0.f, 0.f};

#if LSTM_ASYNC
  // prime step-0 slab
  #pragma unroll
  for (int gt = 0; gt < 4; ++gt)
    async_ld_b64(xgrow + gt * 256, &sbuf[0][eb][gt * 256 + ej]);
#endif

  for (int s = 0; s < LSTM_S; ++s) {
    __syncthreads();  // hbuf ready for this step

    // ---- g = h @ U_all (this WG's 16 batch rows, all 1024 gate cols) ----
    v8f acc0 = zero8, acc1 = zero8;
    #pragma unroll
    for (int kt = 0; kt < 8; ++kt) {
      const __bf16* hp = &hbuf[lrow][kt * 32 + lhi * 8];
      FragU au;
      au.u[0] = *(const u32x4*)(hp);
      au.u[1] = *(const u32x4*)(hp + 16);
      acc0 = __builtin_amdgcn_wmma_f32_16x16x32_bf16(
          false, au.v, false, Bf[0][kt], (short)0, acc0, false, false);
      acc1 = __builtin_amdgcn_wmma_f32_16x16x32_bf16(
          false, au.v, false, Bf[1][kt], (short)0, acc1, false, false);
    }
    #pragma unroll
    for (int r = 0; r < 8; ++r) {
      gbuf[r + lhi * 8][(wid * 2 + 0) * 16 + lrow] = acc0[r];
      gbuf[r + lhi * 8][(wid * 2 + 1) * 16 + lrow] = acc1[r];
    }
    __syncthreads();  // gbuf complete

    // ---- gates + cell update (thread-owned C, h -> LDS bf16) ----
    {
      v4bf xf, xi, xo, xc;
#if LSTM_ASYNC
      // issue step s+1 slab, then drain step s slab (4 loads stay in flight)
      const int cur = s & 1;
      if (s + 1 < LSTM_S) {
        const __bf16* xn = xgrow + (size_t)(s + 1) * (LSTM_B * LSTM_G);
        #pragma unroll
        for (int gt = 0; gt < 4; ++gt)
          async_ld_b64(xn + gt * 256, &sbuf[cur ^ 1][eb][gt * 256 + ej]);
        wait_async4();
      } else {
        wait_async0();
      }
      const __bf16* sb = &sbuf[cur][eb][ej];
      xf = *(const v4bf*)(sb);
      xi = *(const v4bf*)(sb + 256);
      xo = *(const v4bf*)(sb + 512);
      xc = *(const v4bf*)(sb + 768);
#else
      const __bf16* xr = xgrow + (size_t)s * (LSTM_B * LSTM_G);
      xf = *(const v4bf*)(xr);
      xi = *(const v4bf*)(xr + 256);
      xo = *(const v4bf*)(xr + 512);
      xc = *(const v4bf*)(xr + 768);
      if (s + 1 < LSTM_S)
        __builtin_prefetch((const void*)(xr + LSTM_B * LSTM_G), 0, 0);
#endif
      f32x4 gF4 = *(const f32x4*)&gbuf[eb][      ej];
      f32x4 gI4 = *(const f32x4*)&gbuf[eb][256 + ej];
      f32x4 gO4 = *(const f32x4*)&gbuf[eb][512 + ej];
      f32x4 gC4 = *(const f32x4*)&gbuf[eb][768 + ej];

      v4bf hv;
      #pragma unroll
      for (int q = 0; q < 4; ++q) {
        float fg = fast_sigmoid(gF4[q] + (float)xf[q]);
        float ig = fast_sigmoid(gI4[q] + (float)xi[q]);
        float og = fast_sigmoid(gO4[q] + (float)xo[q]);
        float cg = fast_sigmoid(gC4[q] + (float)xc[q]);  // sigmoid candidate (per ref)
        Creg[q] = fg * Creg[q] + ig * cg;
        hv[q] = (__bf16)(og * fast_tanh(Creg[q]));
      }
      *(v4bf*)&hbuf[eb][ej] = hv;
    }
  }

  __syncthreads();  // final h_T in hbuf

  // ---- epilogue: out = h_T @ Why + bias_y (waves 0..15, one N-tile each) ----
  if (wid < 16) {
    v8f acc = zero8;
    #pragma unroll
    for (int kt = 0; kt < 8; ++kt) {
      const __bf16* hp = &hbuf[lrow][kt * 32 + lhi * 8];
      FragU au;
      au.u[0] = *(const u32x4*)(hp);
      au.u[1] = *(const u32x4*)(hp + 16);
      const __bf16* p = Whyt + (size_t)(wid * 16 + lrow) * 256 + kt * 32 + lhi * 8;
      FragU bu;
      bu.u[0] = *(const u32x4*)(p);
      bu.u[1] = *(const u32x4*)(p + 16);
      acc = __builtin_amdgcn_wmma_f32_16x16x32_bf16(
          false, au.v, false, bu.v, (short)0, acc, false, false);
    }
    const float byv = b_y[wid * 16 + lrow];
    #pragma unroll
    for (int r = 0; r < 8; ++r) {
      const int grow = bbase + r + lhi * 8;
      out[(size_t)grow * 256 + wid * 16 + lrow] = acc[r] + byv;
    }
  }
}

// ---------------------------------------------------------------------------
extern "C" void kernel_launch(void* const* d_in, const int* in_sizes, int n_in,
                              void* d_out, int out_size, void* d_ws, size_t ws_size,
                              hipStream_t stream) {
  (void)in_sizes; (void)n_in; (void)out_size; (void)ws_size;
  const float* x   = (const float*)d_in[0];
  const float* Wf  = (const float*)d_in[1];
  const float* Uf  = (const float*)d_in[2];
  const float* bfp = (const float*)d_in[3];
  const float* Wi  = (const float*)d_in[4];
  const float* Ui  = (const float*)d_in[5];
  const float* bip = (const float*)d_in[6];
  const float* Wc  = (const float*)d_in[7];
  const float* Uc  = (const float*)d_in[8];
  const float* bcp = (const float*)d_in[9];
  const float* Wo  = (const float*)d_in[10];
  const float* Uo  = (const float*)d_in[11];
  const float* bop = (const float*)d_in[12];
  const float* Why = (const float*)d_in[13];
  const float* byp = (const float*)d_in[14];

  char* ws = (char*)d_ws;
  size_t off = 0;
  __bf16* xg   = (__bf16*)(ws + off); off += (size_t)LSTM_S * LSTM_B * LSTM_G * 2; // 512 MB
  __bf16* Wt   = (__bf16*)(ws + off); off += (size_t)1024 * 256 * 2;
  __bf16* Utb  = (__bf16*)(ws + off); off += (size_t)1024 * 256 * 2;
  __bf16* Whyt = (__bf16*)(ws + off);

  lstm_cvt_weights<<<2304, 256, 0, stream>>>(Wf, Wi, Wo, Wc, Uf, Ui, Uo, Uc, Why,
                                             Wt, Utb, Whyt);
  lstm_xproj<<<32768, 256, 0, stream>>>(x, Wt, bfp, bip, bop, bcp, xg);
  lstm_scan<<<4, 1024, 0, stream>>>(xg, Utb, Whyt, byp, (float*)d_out);
}
